// VectorQuantizerEMA_11235634447056
// MI455X (gfx1250) — matrix operand order repaired and checked by reading, not compiled, not run on hardware
//
#include <hip/hip_runtime.h>

// ---------------------------------------------------------------------------
// VQ-VAE vector quantizer for MI455X (gfx1250, wave32, WMMA).
// inputs:  d_in[0] = x  [2,256,24,24,24] fp32   (7,077,888 elems)
//          d_in[1] = W  [2048,256]       fp32   (524,288 elems)
// outputs (concat, fp32): loss(1) | quantized(7,077,888) | perplexity(1) |
//          encodings(27648*2048) | indices(27648)
// ---------------------------------------------------------------------------

typedef __attribute__((ext_vector_type(16))) __bf16        v16bf;
typedef __attribute__((ext_vector_type(8)))  float         v8f;
typedef __attribute__((ext_vector_type(4)))  unsigned int  u32x4;
typedef __attribute__((ext_vector_type(2)))  float         f32x2;

constexpr int BATCH   = 2;
constexpr int DIM     = 256;
constexpr int SPATIAL = 24 * 24 * 24;          // 13824
constexpr int NVEC    = BATCH * SPATIAL;       // 27648
constexpr int KEMB    = 2048;

// d_out offsets (floats)
constexpr size_t QUANT_OFF = 1;
constexpr size_t PERP_OFF  = 1 + (size_t)BATCH * DIM * SPATIAL;       // 7,077,889
constexpr size_t ENC_OFF   = PERP_OFF + 1;                            // 7,077,890
constexpr size_t IDX_OFF   = ENC_OFF + (size_t)NVEC * KEMB;           // 63,700,994

// workspace offsets (bytes, all 256B aligned)
constexpr size_t OFF_WHI   = 0;                                // 2048*256*2 = 1,048,576
constexpr size_t OFF_WLO   = OFF_WHI + (size_t)KEMB * DIM * 2;
constexpr size_t OFF_WNORM = OFF_WLO + (size_t)KEMB * DIM * 2; // 8192 B
constexpr size_t OFF_XHI   = OFF_WNORM + (size_t)KEMB * 4;
constexpr size_t OFF_XLO   = OFF_XHI + (size_t)NVEC * DIM * 2; // 14,155,776 each
constexpr size_t OFF_BEST  = OFF_XLO + (size_t)NVEC * DIM * 2;
constexpr size_t OFF_CNT   = OFF_BEST + (size_t)NVEC * 4;
constexpr size_t OFF_SSE   = OFF_CNT + (size_t)KEMB * 4;

union Frag { u32x4 q[2]; v16bf v; };

static __device__ inline v8f wmma_bf16(v16bf a, v16bf b, v8f c) {
  // D = A(16x32 bf16) * B(32x16 bf16) + C(16x16 f32)
  return __builtin_amdgcn_wmma_f32_16x16x32_bf16(
      /*neg_a=*/false, a, /*neg_b=*/false, b,
      /*c_mod=*/(short)0, c, /*reuse_a=*/false, /*reuse_b=*/false);
}

// ---------------------------------------------------------------------------
__global__ __launch_bounds__(256) void vq_init(int* __restrict__ counts,
                                               float* __restrict__ sse) {
  int t = threadIdx.x;
  for (int k = t; k < KEMB; k += 256) counts[k] = 0;
  if (t == 0) *sse = 0.0f;
}

// Split codebook into bf16 hi/lo and compute ||w_k||^2.
__global__ __launch_bounds__(256) void vq_prep_w(const float* __restrict__ W,
                                                 __bf16* __restrict__ whi,
                                                 __bf16* __restrict__ wlo,
                                                 float* __restrict__ wnorm) {
  int k = blockIdx.x, d = threadIdx.x;
  float v  = W[(size_t)k * DIM + d];
  __bf16 h = (__bf16)v;
  __bf16 l = (__bf16)(v - (float)h);
  whi[(size_t)k * DIM + d] = h;
  wlo[(size_t)k * DIM + d] = l;

  float sq = v * v;
  for (int off = 16; off >= 1; off >>= 1) sq += __shfl_xor(sq, off, 32);
  __shared__ float red[8];
  if ((threadIdx.x & 31) == 0) red[threadIdx.x >> 5] = sq;
  __syncthreads();
  if (threadIdx.x == 0) {
    float s = 0.f;
#pragma unroll
    for (int i = 0; i < 8; ++i) s += red[i];
    wnorm[k] = s;
  }
}

// [B,C,S] -> row-major [N,D] bf16 hi/lo via 32x32 LDS tile transpose.
__global__ __launch_bounds__(256) void vq_prep_x(const float* __restrict__ in,
                                                 __bf16* __restrict__ xhi,
                                                 __bf16* __restrict__ xlo) {
  __shared__ float tile[32][33];
  int blk = blockIdx.x;
  int dt = blk & 7;                 // 8 d-tiles
  int st = (blk >> 3) % (SPATIAL / 32);
  int b  = blk / (8 * (SPATIAL / 32));
  int tx = threadIdx.x & 31, ty = threadIdx.x >> 5;

#pragma unroll
  for (int i = 0; i < 4; ++i) {
    int d = dt * 32 + ty + i * 8;
    int s = st * 32 + tx;
    tile[ty + i * 8][tx] = in[((size_t)(b * DIM + d)) * SPATIAL + s];  // coalesced in s
  }
  __syncthreads();
#pragma unroll
  for (int i = 0; i < 4; ++i) {
    int sl = ty + i * 8;
    int n  = b * SPATIAL + st * 32 + sl;
    int d  = dt * 32 + tx;
    float v  = tile[tx][sl];
    __bf16 h = (__bf16)v;
    __bf16 l = (__bf16)(v - (float)h);
    xhi[(size_t)n * DIM + d] = h;                                      // coalesced in d
    xlo[(size_t)n * DIM + d] = l;
  }
}

// ---------------------------------------------------------------------------
// WMMA distance GEMM + argmin.  1 wave = 16 rows; block = 8 waves = 128 rows.
// score(n,k) = ||w_k||^2 - 2 * x_n . w_k   (||x||^2 constant per row)
// dot via bf16 hi/lo split: hi*hi + hi*lo + lo*hi  (~fp32 accurate).
// W k-tiles are staged in double-buffered LDS (shared by all 8 waves; 8x less
// L2 traffic), software-pipelined: next tile's global loads are issued before
// the current tile's 24 WMMAs, parked in registers, stored to LDS after.
// ---------------------------------------------------------------------------
constexpr int LDS_STRIDE = 264;   // bf16 elems per row (padded: bank step 4)

__global__ __launch_bounds__(256) void vq_gemm_argmin(
    const __bf16* __restrict__ xhi, const __bf16* __restrict__ xlo,
    const __bf16* __restrict__ whi, const __bf16* __restrict__ wlo,
    const float* __restrict__ wnorm, int* __restrict__ bestOut,
    int* __restrict__ counts, float* __restrict__ idxOut) {
  __shared__ alignas(16) __bf16 sw[2][2][16 * LDS_STRIDE];   // ~33 KB

  const int t    = threadIdx.x;
  const int lane = t & 31;
  const int wave = t >> 5;
  const int half = lane >> 4;            // 0: lanes 0-15, 1: lanes 16-31
  const int n0   = blockIdx.x * 128 + wave * 16;
  const int row  = n0 + (lane & 15);     // A-matrix row for this lane

  // Staging role of this thread: row sr (0..15), 16-elem chunk at col sc.
  const int sr = t >> 4;
  const int sc = (t & 15) * 16;

  // Resident A fragments: 8 chunks of K=32 over D=256, hi and lo.
  v16bf ahi[8], alo[8];
  {
    const __bf16* rh = xhi + (size_t)row * DIM;
    const __bf16* rl = xlo + (size_t)row * DIM;
#pragma unroll
    for (int c = 0; c < 8; ++c) {
      int d0 = c * 32 + half * 8;        // A layout: elems[0..7]=d0.., [8..15]=d0+16..
      Frag fh, fl;
      fh.q[0] = *(const u32x4*)(rh + d0);
      fh.q[1] = *(const u32x4*)(rh + d0 + 16);
      fl.q[0] = *(const u32x4*)(rl + d0);
      fl.q[1] = *(const u32x4*)(rl + d0 + 16);
      ahi[c] = fh.v;
      alo[c] = fl.v;
    }
  }

  // Prologue: stage k-tile 0 into buffer 0.
  {
    const __bf16* ph = whi + (size_t)sr * DIM + sc;
    const __bf16* pl = wlo + (size_t)sr * DIM + sc;
    __bf16* dh = &sw[0][0][sr * LDS_STRIDE + sc];
    __bf16* dl = &sw[0][1][sr * LDS_STRIDE + sc];
    ((u32x4*)dh)[0] = ((const u32x4*)ph)[0];
    *(u32x4*)(dh + 8) = ((const u32x4*)ph)[1];
    ((u32x4*)dl)[0] = ((const u32x4*)pl)[0];
    *(u32x4*)(dl + 8) = ((const u32x4*)pl)[1];
  }
  __syncthreads();

  float bestd[8];
  int   besti[8];
#pragma unroll
  for (int r = 0; r < 8; ++r) { bestd[r] = 3.4e38f; besti[r] = 0; }

  for (int k0 = 0; k0 < KEMB; k0 += 16) {
    const int  buf      = (k0 >> 4) & 1;
    const bool haveNext = (k0 + 16) < KEMB;      // wave-uniform

    // 1) Issue next tile's global loads; park in registers.
    u32x4 nh0{}, nh1{}, nl0{}, nl1{};
    if (haveNext) {
      const __bf16* ph = whi + (size_t)(k0 + 16 + sr) * DIM + sc;
      const __bf16* pl = wlo + (size_t)(k0 + 16 + sr) * DIM + sc;
      nh0 = ((const u32x4*)ph)[0];
      nh1 = ((const u32x4*)ph)[1];
      nl0 = ((const u32x4*)pl)[0];
      nl1 = ((const u32x4*)pl)[1];
    }

    // 2) Compute current tile from LDS.
    const __bf16* sh = sw[buf][0];
    const __bf16* sl = sw[buf][1];
    const int     kcol = k0 + (lane & 15);       // B-matrix column for this lane
    v8f acc0 = {0, 0, 0, 0, 0, 0, 0, 0};        // hi*hi chain
    v8f acc1 = {0, 0, 0, 0, 0, 0, 0, 0};        // cross-term chain
#pragma unroll
    for (int c = 0; c < 8; ++c) {
      int off = (lane & 15) * LDS_STRIDE + c * 32 + half * 8;
      Frag bh, bl;
      bh.q[0] = *(const u32x4*)(sh + off);       // ds_load_b128, conflict-free
      bh.q[1] = *(const u32x4*)(sh + off + 16);
      bl.q[0] = *(const u32x4*)(sl + off);
      bl.q[1] = *(const u32x4*)(sl + off + 16);
      acc0 = wmma_bf16(ahi[c], bh.v, acc0);
      acc1 = wmma_bf16(ahi[c], bl.v, acc1);
      acc1 = wmma_bf16(alo[c], bh.v, acc1);
    }
    const float wn = wnorm[kcol];
#pragma unroll
    for (int r = 0; r < 8; ++r) {
      float dist = wn - 2.0f * (acc0[r] + acc1[r]);
      if (dist < bestd[r]) { bestd[r] = dist; besti[r] = kcol; }
    }

    // 3) Commit staged registers to the other LDS buffer.
    if (haveNext) {
      __bf16* dh = &sw[buf ^ 1][0][sr * LDS_STRIDE + sc];
      __bf16* dl = &sw[buf ^ 1][1][sr * LDS_STRIDE + sc];
      ((u32x4*)dh)[0] = nh0;
      *(u32x4*)(dh + 8) = nh1;
      ((u32x4*)dl)[0] = nl0;
      *(u32x4*)(dl + 8) = nl1;
    }
    __syncthreads();
  }

  // Reduce argmin across the 16 lanes sharing each row (xor stays in half).
#pragma unroll
  for (int r = 0; r < 8; ++r) {
    for (int off = 8; off >= 1; off >>= 1) {
      float od = __shfl_xor(bestd[r], off, 32);
      int   oi = __shfl_xor(besti[r], off, 32);
      if (od < bestd[r] || (od == bestd[r] && oi < besti[r])) {
        bestd[r] = od; besti[r] = oi;
      }
    }
  }
  if ((lane & 15) == 0) {
#pragma unroll
    for (int r = 0; r < 8; ++r) {
      int n  = n0 + half * 8 + r;        // C/D layout: m = r + 8*(lane/16)
      int bi = besti[r];
      bestOut[n] = bi;
      idxOut[n]  = (float)bi;
      atomicAdd(&counts[bi], 1);
    }
  }
}

// Gather quantized rows back to [B,C,S] and accumulate SSE for the loss.
__global__ __launch_bounds__(256) void vq_quant_loss(
    const float* __restrict__ in, const float* __restrict__ W,
    const int* __restrict__ bestIdx, float* __restrict__ qout,
    float* __restrict__ sse) {
  size_t e = (size_t)blockIdx.x * 256 + threadIdx.x;  // < B*DIM*SPATIAL exactly
  int s = (int)(e % SPATIAL);
  int d = (int)((e / SPATIAL) & (DIM - 1));
  int b = (int)(e / ((size_t)SPATIAL * DIM));
  int n = b * SPATIAL + s;
  int idx  = bestIdx[n];
  float q  = W[(size_t)idx * DIM + d];   // 2MB codebook: L2 resident
  float xv = in[e];
  qout[e] = q;                           // straight-through value == quantized
  float df = q - xv;
  float sq = df * df;
  for (int off = 16; off >= 1; off >>= 1) sq += __shfl_xor(sq, off, 32);
  __shared__ float red[8];
  if ((threadIdx.x & 31) == 0) red[threadIdx.x >> 5] = sq;
  __syncthreads();
  if (threadIdx.x == 0) {
    float tot = 0.f;
#pragma unroll
    for (int i = 0; i < 8; ++i) tot += red[i];
    atomicAdd(sse, tot);
  }
}

// One-hot encodings: 226 MB of writes -> nontemporal stores (don't pollute L2).
// float2 stores: ENC_OFF is only 8B-aligned in d_out.
__global__ __launch_bounds__(256) void vq_encodings(const int* __restrict__ bestIdx,
                                                    float* __restrict__ enc) {
  int n = blockIdx.x, t = threadIdx.x;
  int idx = bestIdx[n];
  size_t base = (size_t)n * KEMB + (size_t)t * 8;
  f32x2 p[4] = {{0.f, 0.f}, {0.f, 0.f}, {0.f, 0.f}, {0.f, 0.f}};
  int rel = idx - t * 8;
  if (rel >= 0 && rel < 8) p[rel >> 1][rel & 1] = 1.0f;
#pragma unroll
  for (int j = 0; j < 4; ++j)
    __builtin_nontemporal_store(p[j], (f32x2*)(enc + base + j * 2));
}

__global__ __launch_bounds__(256) void vq_finalize(const int* __restrict__ counts,
                                                   const float* __restrict__ sse,
                                                   float* __restrict__ out) {
  __shared__ float red[256];
  float partial = 0.f;
  const float invN = 1.0f / (float)NVEC;
  for (int k = threadIdx.x; k < KEMB; k += 256) {
    float p = (float)counts[k] * invN;
    partial += p * __logf(p + 1e-10f);
  }
  red[threadIdx.x] = partial;
  __syncthreads();
  for (int s = 128; s > 0; s >>= 1) {
    if (threadIdx.x < s) red[threadIdx.x] += red[threadIdx.x + s];
    __syncthreads();
  }
  if (threadIdx.x == 0) {
    out[PERP_OFF] = __expf(-red[0]);
    out[0] = 0.25f * (*sse) / (float)((size_t)BATCH * DIM * SPATIAL);
  }
}

// ---------------------------------------------------------------------------
extern "C" void kernel_launch(void* const* d_in, const int* in_sizes, int n_in,
                              void* d_out, int out_size, void* d_ws, size_t ws_size,
                              hipStream_t stream) {
  (void)in_sizes; (void)n_in; (void)out_size; (void)ws_size;
  const float* x_in = (const float*)d_in[0];
  const float* W    = (const float*)d_in[1];
  float* out = (float*)d_out;
  char*  ws  = (char*)d_ws;

  __bf16* whi   = (__bf16*)(ws + OFF_WHI);
  __bf16* wlo   = (__bf16*)(ws + OFF_WLO);
  float*  wnorm = (float*)(ws + OFF_WNORM);
  __bf16* xhi   = (__bf16*)(ws + OFF_XHI);
  __bf16* xlo   = (__bf16*)(ws + OFF_XLO);
  int*    best  = (int*)(ws + OFF_BEST);
  int*    cnt   = (int*)(ws + OFF_CNT);
  float*  sse   = (float*)(ws + OFF_SSE);

  vq_init<<<1, 256, 0, stream>>>(cnt, sse);
  vq_prep_w<<<KEMB, 256, 0, stream>>>(W, whi, wlo, wnorm);
  vq_prep_x<<<BATCH * (SPATIAL / 32) * (DIM / 32), 256, 0, stream>>>(x_in, xhi, xlo);
  vq_gemm_argmin<<<NVEC / 128, 256, 0, stream>>>(xhi, xlo, whi, wlo, wnorm,
                                                 best, cnt, out + IDX_OFF);
  vq_quant_loss<<<(BATCH * DIM * SPATIAL) / 256, 256, 0, stream>>>(x_in, W, best,
                                                                  out + QUANT_OFF, sse);
  vq_encodings<<<NVEC, 256, 0, stream>>>(best, out + ENC_OFF);
  vq_finalize<<<1, 256, 0, stream>>>(cnt, sse, out);
}
